// LstmAttentionNet_49967649522002
// MI455X (gfx1250) — matrix operand order repaired and checked by reading, not compiled, not run on hardware
//
#include <hip/hip_runtime.h>
#include <hip/hip_bf16.h>
#include <stdint.h>

typedef __attribute__((ext_vector_type(16))) _Float16 v16h;
typedef __attribute__((ext_vector_type(8)))  _Float16 v8h;
typedef __attribute__((ext_vector_type(8)))  float    v8f;

#define WMMA_F16(a,b,c) __builtin_amdgcn_wmma_f32_16x16x32_f16(false,(a),false,(b),(short)0,(c),false,false)

__device__ __forceinline__ float sigmoidf_(float x) { return 1.f / (1.f + __expf(-x)); }
__device__ __forceinline__ float tanhf_(float x)    { return 1.f - 2.f / (__expf(2.f * x) + 1.f); }

// A-tile 16x32 f16 from row-major storage (row stride lda).
// lanes 0-15: row=l,   halves[0..7]=K kb..kb+7,   halves[8..15]=kb+16..kb+23
// lanes16-31: row=l-16,halves[0..7]=K kb+8..kb+15,halves[8..15]=kb+24..kb+31
__device__ __forceinline__ v16h ld_tile_a(const _Float16* base, int lda, int kb, int lane) {
  const _Float16* p = base + (size_t)(lane & 15) * lda + kb + ((lane & 16) ? 8 : 0);
  v8h lo = *(const v8h*)(p);
  v8h hi = *(const v8h*)(p + 16);
  v16h r;
#pragma unroll
  for (int i = 0; i < 8; ++i) { r[i] = lo[i]; r[i + 8] = hi[i]; }
  return r;
}

// B-tile 32x16 (KxN) stored as (N,K) row-major (row stride ldb).
// lanes 0-15: col=l,    halves[0..7]=K kb..kb+7,   halves[8..15]=kb+16..kb+23
// lanes16-31: col=l-16, halves[0..7]=K kb+8..kb+15,halves[8..15]=kb+24..kb+31
__device__ __forceinline__ v16h ld_tile_b(const _Float16* base, int ldb, int kb, int lane) {
  const _Float16* p = base + (size_t)(lane & 15) * ldb + kb + ((lane & 16) ? 8 : 0);
  v8h lo = *(const v8h*)(p);
  v8h hi = *(const v8h*)(p + 16);
  v16h r;
#pragma unroll
  for (int i = 0; i < 8; ++i) { r[i] = lo[i]; r[i + 8] = hi[i]; }
  return r;
}

// ---------------------------------------------------------------------------
// Persistent-workgroup LSTM: one 1024-thread block per (instance, direction).
// h state (64x384) kept in LDS as f16; c state in accumulator-layout VGPRs.
// Per step: gates = bias + x_t @ Wih^T + h @ Whh^T via WMMA (f16 in, f32 acc).
// ---------------------------------------------------------------------------
__global__ void __launch_bounds__(1024)
k_lstm(const _Float16* __restrict__ x,
       const _Float16* __restrict__ WihF, const _Float16* __restrict__ WhhF, const float* __restrict__ biasF,
       const _Float16* __restrict__ WihB, const _Float16* __restrict__ WhhB, const float* __restrict__ biasB,
       float* __restrict__ hout, _Float16* __restrict__ hout16,
       int T, int Dx, int ldh,
       long long xInstStride, long long hInstStride)
{
  const int H = 384, NB = 64;
  const int dir  = blockIdx.x & 1;
  const int inst = blockIdx.x >> 1;
  x    += (long long)inst * xInstStride;
  hout += (long long)inst * hInstStride;
  if (hout16) hout16 += (long long)inst * hInstStride;
  const _Float16* Wih  = dir ? WihB  : WihF;
  const _Float16* Whh  = dir ? WhhB  : WhhF;
  const float*    bias = dir ? biasB : biasF;
  const int colOff = dir * H;

  __shared__ __align__(16) _Float16 hlds[64 * 384];
  for (int i = threadIdx.x; i < 64 * 384; i += blockDim.x) hlds[i] = (_Float16)0.0f;
  __syncthreads();

  const int wave = threadIdx.x >> 5;
  const int lane = threadIdx.x & 31;

  float cst[3][8];
#pragma unroll
  for (int g = 0; g < 3; ++g)
#pragma unroll
    for (int v = 0; v < 8; ++v) cst[g][v] = 0.f;

  for (int s = 0; s < T; ++s) {
    const int t = dir ? (T - 1 - s) : s;
    const _Float16* xt = x + (long long)t * NB * Dx;
    float hvals[3][8];

#pragma unroll 1
    for (int g3 = 0; g3 < 3; ++g3) {
      const int tg = wave + 32 * g3;       // 96 tile-groups: 4 row-tiles x 24 col-tiles
      const int rt = tg / 24;
      const int nt = tg % 24;
      v8f acc0, acc1, acc2, acc3;
      {
        const int cb = nt * 16 + (lane & 15);
        float b0 = bias[0 * H + cb], b1 = bias[1 * H + cb];
        float b2 = bias[2 * H + cb], b3 = bias[3 * H + cb];
#pragma unroll
        for (int v = 0; v < 8; ++v) { acc0[v] = b0; acc1[v] = b1; acc2[v] = b2; acc3[v] = b3; }
      }
      // x @ Wih^T
      const _Float16* aBaseX = xt + (long long)(rt * 16) * Dx;
      for (int kb = 0; kb < Dx; kb += 32) {
        v16h a = ld_tile_a(aBaseX, Dx, kb, lane);
        v16h b0 = ld_tile_b(Wih + (long long)(0 * H + nt * 16) * Dx, Dx, kb, lane);
        acc0 = WMMA_F16(a, b0, acc0);
        v16h b1 = ld_tile_b(Wih + (long long)(1 * H + nt * 16) * Dx, Dx, kb, lane);
        acc1 = WMMA_F16(a, b1, acc1);
        v16h b2 = ld_tile_b(Wih + (long long)(2 * H + nt * 16) * Dx, Dx, kb, lane);
        acc2 = WMMA_F16(a, b2, acc2);
        v16h b3 = ld_tile_b(Wih + (long long)(3 * H + nt * 16) * Dx, Dx, kb, lane);
        acc3 = WMMA_F16(a, b3, acc3);
      }
      // h @ Whh^T (h from LDS, f16)
      const _Float16* aBaseH = &hlds[(rt * 16) * 384];
      for (int kb = 0; kb < 384; kb += 32) {
        v16h a = ld_tile_a(aBaseH, 384, kb, lane);
        v16h b0 = ld_tile_b(Whh + (long long)(0 * H + nt * 16) * 384, 384, kb, lane);
        acc0 = WMMA_F16(a, b0, acc0);
        v16h b1 = ld_tile_b(Whh + (long long)(1 * H + nt * 16) * 384, 384, kb, lane);
        acc1 = WMMA_F16(a, b1, acc1);
        v16h b2 = ld_tile_b(Whh + (long long)(2 * H + nt * 16) * 384, 384, kb, lane);
        acc2 = WMMA_F16(a, b2, acc2);
        v16h b3 = ld_tile_b(Whh + (long long)(3 * H + nt * 16) * 384, 384, kb, lane);
        acc3 = WMMA_F16(a, b3, acc3);
      }
#pragma unroll
      for (int v = 0; v < 8; ++v) {
        float iv = sigmoidf_(acc0[v]);
        float fv = sigmoidf_(acc1[v]);
        float gv = tanhf_(acc2[v]);
        float ov = sigmoidf_(acc3[v]);
        float cn = fv * cst[g3][v] + iv * gv;
        cst[g3][v] = cn;
        hvals[g3][v] = ov * tanhf_(cn);
      }
    }
    __syncthreads();   // all waves finished reading old h
#pragma unroll
    for (int g3 = 0; g3 < 3; ++g3) {
      const int tg = wave + 32 * g3;
      const int rt = tg / 24;
      const int nt = tg % 24;
      const int coln = nt * 16 + (lane & 15);
#pragma unroll
      for (int v = 0; v < 8; ++v) {
        const int row = rt * 16 + ((lane & 16) ? (v + 8) : v);
        float hv = hvals[g3][v];
        hlds[row * 384 + coln] = (_Float16)hv;
        long long oo = ((long long)t * NB + row) * ldh + colOff + coln;
        hout[oo] = hv;
        if (hout16) hout16[oo] = (_Float16)hv;
      }
    }
    __syncthreads();   // new h visible before next step
  }
}

// ---------------------------------------------------------------------------
// Generic batched WMMA GEMM: C[z] = A[z] @ B[z]^T (+bias) (* mul), f32/f16 out
// One 16x16 C tile per wave, 8 waves per block. z = z1*zDiv + z2 offsets.
// ---------------------------------------------------------------------------
__global__ void __launch_bounds__(256)
k_gemm(const _Float16* __restrict__ A, long long aHi, long long aLo, int lda,
       const _Float16* __restrict__ Bm, long long bHi, long long bLo, int ldb,
       float* __restrict__ Cf, _Float16* __restrict__ Ch, long long cHi, long long cLo, int ldc,
       const float* __restrict__ bias,
       const float* __restrict__ mul, long long mHi, long long mLo, int ldm,
       int K, int zDiv, int nTilesN, int tilesPerZ, int totalTiles)
{
  const int gtile = blockIdx.x * 8 + (threadIdx.x >> 5);
  if (gtile >= totalTiles) return;               // wave-uniform guard
  const int lane = threadIdx.x & 31;
  const int z  = gtile / tilesPerZ;
  const int tt = gtile % tilesPerZ;
  const int tm = tt / nTilesN;
  const int tn = tt % nTilesN;
  const int z1 = z / zDiv;
  const int z2 = z % zDiv;
  const _Float16* Ab = A  + (long long)z1 * aHi + (long long)z2 * aLo + (long long)(tm * 16) * lda;
  const _Float16* Bb = Bm + (long long)z1 * bHi + (long long)z2 * bLo + (long long)(tn * 16) * ldb;
  v8f acc;
#pragma unroll
  for (int v = 0; v < 8; ++v) acc[v] = 0.f;
  for (int kb = 0; kb < K; kb += 32) {
    v16h a = ld_tile_a(Ab, lda, kb, lane);
    v16h b = ld_tile_b(Bb, ldb, kb, lane);
    acc = WMMA_F16(a, b, acc);
  }
  const int coln = tn * 16 + (lane & 15);
  const float bv = bias ? bias[coln] : 0.f;
#pragma unroll
  for (int v = 0; v < 8; ++v) {
    const int row = tm * 16 + ((lane & 16) ? (v + 8) : v);
    float r = acc[v] + bv;
    if (mul) r *= mul[(long long)z1 * mHi + (long long)z2 * mLo + (long long)row * ldm + coln];
    long long off = (long long)z1 * cHi + (long long)z2 * cLo + (long long)row * ldc + coln;
    if (Cf) Cf[off] = r; else Ch[off] = (_Float16)r;
  }
}

// ------------------------------- elementwise -------------------------------
__global__ void k_f32_to_f16(const float* __restrict__ s, _Float16* __restrict__ d, int n) {
  int i = blockIdx.x * blockDim.x + threadIdx.x;
  if (i < n) d[i] = (_Float16)s[i];
}

// context (B,Tc,E) f32 -> (Tc,B,E) f16
__global__ void k_prep_context(const float* __restrict__ src, _Float16* __restrict__ dst, int total) {
  int idx = blockIdx.x * blockDim.x + threadIdx.x;
  if (idx >= total) return;
  int e = idx % 384; int r = idx / 384; int b = r % 64; int t = r / 64;
  dst[idx] = (_Float16)src[((long long)b * 384 + t) * 384 + e];
}

// options (B,No,To,381) f32 -> (No,To,B,384) f16, pad last 3 with 0
__global__ void k_prep_options(const float* __restrict__ src, _Float16* __restrict__ dst, int total) {
  int idx = blockIdx.x * blockDim.x + threadIdx.x;
  if (idx >= total) return;
  int e = idx % 384; int r = idx / 384; int b = r % 64; r /= 64; int t = r % 64; int opt = r / 64;
  float v = (e < 381) ? src[(((long long)b * 5 + opt) * 64 + t) * 381 + e] : 0.f;
  dst[idx] = (_Float16)v;
}

// ctx f32 (Tc,B,F) -> ctx_f16 (same) + ctxT_f16 (B,F,Tc)
__global__ void k_ctx_expand(const float* __restrict__ ctx, _Float16* __restrict__ c16,
                             _Float16* __restrict__ cT16, int total) {
  int idx = blockIdx.x * blockDim.x + threadIdx.x;
  if (idx >= total) return;
  int f = idx % 768; int r = idx / 768; int b = r % 64; int s = r / 64;
  float v = ctx[idx];
  c16[idx] = (_Float16)v;
  cT16[((long long)b * 768 + f) * 384 + s] = (_Float16)v;
}

// row-wise softmax over 384 cols; writes f32 (d_out region) + f16
__global__ void k_softmax(const float* __restrict__ s, float* __restrict__ outf,
                          _Float16* __restrict__ out16) {
  long long row = blockIdx.x;
  const float* src = s + row * 384;
  int tid = threadIdx.x;    // 128 threads, 3 cols each
  float v0 = src[tid], v1 = src[tid + 128], v2 = src[tid + 256];
  __shared__ float red[128];
  red[tid] = fmaxf(v0, fmaxf(v1, v2));
  __syncthreads();
  for (int st = 64; st > 0; st >>= 1) { if (tid < st) red[tid] = fmaxf(red[tid], red[tid + st]); __syncthreads(); }
  float m = red[0]; __syncthreads();
  v0 = __expf(v0 - m); v1 = __expf(v1 - m); v2 = __expf(v2 - m);
  red[tid] = v0 + v1 + v2;
  __syncthreads();
  for (int st = 64; st > 0; st >>= 1) { if (tid < st) red[tid] += red[tid + st]; __syncthreads(); }
  float inv = 1.f / red[0];
  float r0 = v0 * inv, r1 = v1 * inv, r2 = v2 * inv;
  long long o = row * 384;
  outf[o + tid] = r0; outf[o + tid + 128] = r1; outf[o + tid + 256] = r2;
  out16[o + tid] = (_Float16)r0; out16[o + tid + 128] = (_Float16)r1; out16[o + tid + 256] = (_Float16)r2;
}

// att_c = softmax_s( max_t scores[z,t,s] ), one block per z=(opt,b)
__global__ void k_attc(const float* __restrict__ scores, float* __restrict__ attc) {
  int z = blockIdx.x;
  int tid = threadIdx.x;   // 128
  const float* base = scores + (long long)z * 64 * 384;
  float v[3];
#pragma unroll
  for (int j = 0; j < 3; ++j) {
    int sc = tid + 128 * j;
    float m = -1e30f;
    for (int t = 0; t < 64; ++t) m = fmaxf(m, base[t * 384 + sc]);
    v[j] = m;
  }
  __shared__ float red[128];
  red[tid] = fmaxf(v[0], fmaxf(v[1], v[2]));
  __syncthreads();
  for (int st = 64; st > 0; st >>= 1) { if (tid < st) red[tid] = fmaxf(red[tid], red[tid + st]); __syncthreads(); }
  float m = red[0]; __syncthreads();
#pragma unroll
  for (int j = 0; j < 3; ++j) v[j] = __expf(v[j] - m);
  red[tid] = v[0] + v[1] + v[2];
  __syncthreads();
  for (int st = 64; st > 0; st >>= 1) { if (tid < st) red[tid] += red[tid + st]; __syncthreads(); }
  float inv = 1.f / red[0];
#pragma unroll
  for (int j = 0; j < 3; ++j) attc[(long long)z * 384 + tid + 128 * j] = v[j] * inv;
}

// mixcont[opt,b,f] = sum_s attc[opt,b,s] * ctx[s,b,f]
__global__ void k_mixcont(const float* __restrict__ attc, const float* __restrict__ ctx,
                          float* __restrict__ mixcont, int total) {
  int idx = blockIdx.x * blockDim.x + threadIdx.x;
  if (idx >= total) return;
  int f = idx % 768; int r = idx / 768; int b = r % 64; int opt = r / 64;
  const float* ac = attc + (long long)(opt * 64 + b) * 384;
  float s = 0.f;
  for (int t = 0; t < 384; ++t) s += ac[t] * ctx[((long long)t * 64 + b) * 768 + f];
  mixcont[idx] = s;
}

// mix[...,0:F] = o ; mix[...,2F:3F] = mixcont * o  (middle third filled by GEMM)
__global__ void k_mix_assemble(const float* __restrict__ o, const float* __restrict__ mixcont,
                               _Float16* __restrict__ mix, int total) {
  int idx = blockIdx.x * blockDim.x + threadIdx.x;
  if (idx >= total) return;
  int f = idx % 768; int r = idx / 768; int b = r % 64; r /= 64; int t = r % 64; int opt = r / 64;
  float ov = o[idx];
  long long mrow = (((long long)opt * 64 + t) * 64 + b) * 2304;
  mix[mrow + f] = (_Float16)ov;
  mix[mrow + 1536 + f] = (_Float16)(ov * mixcont[((long long)opt * 64 + b) * 768 + f]);
}

// h1in[opt,b, 0:768]=ctx[-1]; [768:1536]=mean_t ctx; [1536:2304]=o2[-1]; [2304:3072]=mean_t o2
__global__ void k_build_h1in(const float* __restrict__ ctx, const float* __restrict__ o2,
                             _Float16* __restrict__ h1in, int total) {
  int idx = blockIdx.x * blockDim.x + threadIdx.x;
  if (idx >= total) return;
  int j = idx % 3072; int r = idx / 3072; int b = r % 64; int opt = r / 64;
  float v;
  if (j < 768) {
    v = ctx[((long long)383 * 64 + b) * 768 + j];
  } else if (j < 1536) {
    int f = j - 768; float s = 0.f;
    for (int t = 0; t < 384; ++t) s += ctx[((long long)t * 64 + b) * 768 + f];
    v = s * (1.f / 384.f);
  } else if (j < 2304) {
    int f = j - 1536;
    v = o2[(((long long)opt * 64 + 63) * 64 + b) * 768 + f];
  } else {
    int f = j - 2304; float s = 0.f;
    for (int t = 0; t < 64; ++t) s += o2[(((long long)opt * 64 + t) * 64 + b) * 768 + f];
    v = s * (1.f / 64.f);
  }
  h1in[idx] = (_Float16)v;
}

// training-mode BatchNorm over batch (64) per column + ReLU
__global__ void k_bn(const float* __restrict__ h1, const float* __restrict__ gamma,
                     const float* __restrict__ beta, float* __restrict__ hbn, int total) {
  int idx = blockIdx.x * blockDim.x + threadIdx.x;
  if (idx >= total) return;
  int col = idx % 384; int opt = idx / 384;
  const float* base = h1 + (long long)opt * 64 * 384;
  float m = 0.f;
  for (int b = 0; b < 64; ++b) m += base[b * 384 + col];
  m *= (1.f / 64.f);
  float var = 0.f;
  for (int b = 0; b < 64; ++b) { float d = base[b * 384 + col] - m; var += d * d; }
  var *= (1.f / 64.f);
  float inv = rsqrtf(var + 1e-5f) * gamma[col];
  for (int b = 0; b < 64; ++b) {
    float x = (base[b * 384 + col] - m) * inv + beta[col];
    hbn[(long long)opt * 64 * 384 + b * 384 + col] = fmaxf(x, 0.f);
  }
}

__global__ void k_logits(const float* __restrict__ hbn, const float* __restrict__ W2,
                         const float* __restrict__ b2, float* __restrict__ out, int total) {
  int idx = blockIdx.x * blockDim.x + threadIdx.x;
  if (idx >= total) return;
  int b = idx % 64; int opt = idx / 64;
  const float* h = hbn + ((long long)opt * 64 + b) * 384;
  float s = b2[0];
  for (int k = 0; k < 384; ++k) s += h[k] * W2[k];
  out[b * 5 + opt] = 1.f / (1.f + __expf(-s));
}

// ---------------------------------------------------------------------------
extern "C" void kernel_launch(void* const* d_in, const int* in_sizes, int n_in,
                              void* d_out, int out_size, void* d_ws, size_t ws_size,
                              hipStream_t stream) {
  (void)in_sizes; (void)n_in; (void)out_size; (void)ws_size;
  const int B = 64, Tc = 384, To = 64, No = 5, E = 384, H = 384, F = 768, G4 = 1536, D2 = 2304;

  const float* context  = (const float*)d_in[0];
  const float* options  = (const float*)d_in[2];
  const float* l1WihF = (const float*)d_in[4];
  const float* l1WhhF = (const float*)d_in[5];
  const float* l1bF   = (const float*)d_in[6];
  const float* l1WihB = (const float*)d_in[7];
  const float* l1WhhB = (const float*)d_in[8];
  const float* l1bB   = (const float*)d_in[9];
  const float* l2WihF = (const float*)d_in[10];
  const float* l2WhhF = (const float*)d_in[11];
  const float* l2bF   = (const float*)d_in[12];
  const float* l2WihB = (const float*)d_in[13];
  const float* l2WhhB = (const float*)d_in[14];
  const float* l2bB   = (const float*)d_in[15];
  const float* W_opt  = (const float*)d_in[16];
  const float* b_opt  = (const float*)d_in[17];
  const float* W1     = (const float*)d_in[18];
  const float* b1     = (const float*)d_in[19];
  const float* gamma  = (const float*)d_in[20];
  const float* beta   = (const float*)d_in[21];
  const float* W2     = (const float*)d_in[22];
  const float* b2     = (const float*)d_in[23];
  float* out = (float*)d_out;

  char* ws = (char*)d_ws;
  size_t off = 0;
  auto alloc = [&](size_t bytes) -> void* {
    void* p = ws + off;
    off = (off + bytes + 255) & ~(size_t)255;
    return p;
  };

  // weights f16
  _Float16* w1ihF = (_Float16*)alloc((size_t)G4 * E  * 2);
  _Float16* w1hhF = (_Float16*)alloc((size_t)G4 * H  * 2);
  _Float16* w1ihB = (_Float16*)alloc((size_t)G4 * E  * 2);
  _Float16* w1hhB = (_Float16*)alloc((size_t)G4 * H  * 2);
  _Float16* w2ihF = (_Float16*)alloc((size_t)G4 * D2 * 2);
  _Float16* w2hhF = (_Float16*)alloc((size_t)G4 * H  * 2);
  _Float16* w2ihB = (_Float16*)alloc((size_t)G4 * D2 * 2);
  _Float16* w2hhB = (_Float16*)alloc((size_t)G4 * H  * 2);
  _Float16* wopt16 = (_Float16*)alloc((size_t)F * F * 2);
  _Float16* w1_16  = (_Float16*)alloc((size_t)H * 3072 * 2);
  // activations
  _Float16* ctx16in = (_Float16*)alloc((size_t)Tc * B * E * 2);
  _Float16* opt16   = (_Float16*)alloc((size_t)No * To * B * E * 2);
  float*    ctx     = (float*)   alloc((size_t)Tc * B * F * 4);
  _Float16* ctx16   = (_Float16*)alloc((size_t)Tc * B * F * 2);
  _Float16* ctxT16  = (_Float16*)alloc((size_t)B * F * Tc * 2);
  float*    o       = (float*)   alloc((size_t)No * To * B * F * 4);
  _Float16* o16     = (_Float16*)alloc((size_t)No * To * B * F * 2);
  _Float16* ow16    = (_Float16*)alloc((size_t)No * To * B * F * 2);
  float*    scores  = (float*)   alloc((size_t)No * B * To * Tc * 4);
  _Float16* att16   = (_Float16*)alloc((size_t)No * B * To * Tc * 2);
  float*    attc    = (float*)   alloc((size_t)No * B * Tc * 4);
  float*    mixcont = (float*)   alloc((size_t)No * B * F * 4);
  _Float16* mix16   = (_Float16*)alloc((size_t)No * To * B * D2 * 2);
  float*    o2      = (float*)   alloc((size_t)No * To * B * F * 4);
  _Float16* h1in16  = (_Float16*)alloc((size_t)No * B * 3072 * 2);
  float*    h1      = (float*)   alloc((size_t)No * B * H * 4);
  float*    hbn     = (float*)   alloc((size_t)No * B * H * 4);

  auto cvt = [&](const float* s, _Float16* d, int n) {
    k_f32_to_f16<<<(n + 255) / 256, 256, 0, stream>>>(s, d, n);
  };
  cvt(l1WihF, w1ihF, G4 * E);  cvt(l1WhhF, w1hhF, G4 * H);
  cvt(l1WihB, w1ihB, G4 * E);  cvt(l1WhhB, w1hhB, G4 * H);
  cvt(l2WihF, w2ihF, G4 * D2); cvt(l2WhhF, w2hhF, G4 * H);
  cvt(l2WihB, w2ihB, G4 * D2); cvt(l2WhhB, w2hhB, G4 * H);
  cvt(W_opt, wopt16, F * F);   cvt(W1, w1_16, H * 3072);

  {
    int n = Tc * B * E;
    k_prep_context<<<(n + 255) / 256, 256, 0, stream>>>(context, ctx16in, n);
  }
  {
    int n = No * To * B * E;
    k_prep_options<<<(n + 255) / 256, 256, 0, stream>>>(options, opt16, n);
  }

  // BiLSTM1 over context: 2 workgroups (fwd/bwd)
  k_lstm<<<2, 1024, 0, stream>>>(ctx16in, w1ihF, w1hhF, l1bF, w1ihB, w1hhB, l1bB,
                                 ctx, (_Float16*)nullptr, Tc, E, F, 0LL, 0LL);
  {
    int n = Tc * B * F;
    k_ctx_expand<<<(n + 255) / 256, 256, 0, stream>>>(ctx, ctx16, ctxT16, n);
  }

  // BiLSTM1 over all 5 options: 10 workgroups
  k_lstm<<<10, 1024, 0, stream>>>(opt16, w1ihF, w1hhF, l1bF, w1ihB, w1hhB, l1bB,
                                  o, o16, To, E, F,
                                  (long long)To * B * E, (long long)To * B * F);

  // ow = o @ W_opt^T + b_opt  (M=No*To*B=20480, N=768, K=768) -> f16
  {
    int tiles = (No * To * B / 16) * (F / 16);   // 61440
    k_gemm<<<tiles / 8, 256, 0, stream>>>(o16, 0LL, 0LL, F,
                                          wopt16, 0LL, 0LL, F,
                                          nullptr, ow16, 0LL, 0LL, F,
                                          b_opt, nullptr, 0LL, 0LL, 0,
                                          F, 1, F / 16, tiles, tiles);
  }

  // scores[opt,b,t,s] = sum_f ow[opt,t,b,f]*ctx[s,b,f]; z=(opt,b), M=64,N=384,K=768
  {
    int tilesPerZ = (To / 16) * (Tc / 16);   // 96
    int total = No * B * tilesPerZ;          // 30720
    k_gemm<<<total / 8, 256, 0, stream>>>(ow16, (long long)To * B * F, (long long)F, B * F,
                                          ctx16, 0LL, (long long)F, B * F,
                                          scores, nullptr, (long long)B * To * Tc, (long long)To * Tc, Tc,
                                          nullptr, nullptr, 0LL, 0LL, 0,
                                          F, B, Tc / 16, tilesPerZ, total);
  }

  // att = softmax(scores) -> d_out attention map + f16 copy
  k_softmax<<<No * B * To, 128, 0, stream>>>(scores, out + B * No, att16);

  // att_c and mix_cont
  k_attc<<<No * B, 128, 0, stream>>>(scores, attc);
  {
    int n = No * B * F;
    k_mixcont<<<(n + 255) / 256, 256, 0, stream>>>(attc, ctx, mixcont, n);
  }

  // mix_opt = (att @ ctx) * o -> mix[...,F:2F] f16 ; z=(opt,b), M=64,N=768,K=384
  {
    int tilesPerZ = (To / 16) * (F / 16);    // 192
    int total = No * B * tilesPerZ;          // 61440
    k_gemm<<<total / 8, 256, 0, stream>>>(att16, (long long)B * To * Tc, (long long)To * Tc, Tc,
                                          ctxT16, 0LL, (long long)F * Tc, Tc,
                                          nullptr, mix16 + F, (long long)To * B * D2, (long long)D2, B * D2,
                                          nullptr, o, (long long)To * B * F, (long long)F, B * F,
                                          Tc, B, F / 16, tilesPerZ, total);
  }
  {
    int n = No * To * B * F;
    k_mix_assemble<<<(n + 255) / 256, 256, 0, stream>>>(o, mixcont, mix16, n);
  }

  // BiLSTM2 over mix (Dx=2304): 10 workgroups
  k_lstm<<<10, 1024, 0, stream>>>(mix16, w2ihF, w2hhF, l2bF, w2ihB, w2hhB, l2bB,
                                  o2, (_Float16*)nullptr, To, D2, F,
                                  (long long)To * B * D2, (long long)To * B * F);

  // head
  {
    int n = No * B * 3072;
    k_build_h1in<<<(n + 255) / 256, 256, 0, stream>>>(ctx, o2, h1in16, n);
  }
  {
    int tilesPerZ = (B / 16) * (H / 16);     // 96
    int total = No * tilesPerZ;              // 480
    k_gemm<<<total / 8, 256, 0, stream>>>(h1in16, (long long)B * 3072, 0LL, 3072,
                                          w1_16, 0LL, 0LL, 3072,
                                          h1, nullptr, (long long)B * H, 0LL, H,
                                          b1, nullptr, 0LL, 0LL, 0,
                                          3072, 1, H / 16, tilesPerZ, total);
  }
  {
    int n = No * H;
    k_bn<<<(n + 255) / 256, 256, 0, stream>>>(h1, gamma, beta, hbn, n);
  }
  {
    int n = No * B;
    k_logits<<<(n + 63) / 64, 64, 0, stream>>>(hbn, W2, b2, out, n);
  }
}
